// DiffForest_48558900248870
// MI455X (gfx1250) — matrix operand order repaired and checked by reading, not compiled, not run on hardware
//
#include <hip/hip_runtime.h>
#include <hip/hip_bf16.h>

typedef __attribute__((ext_vector_type(16))) _Float16 v16h;
typedef __attribute__((ext_vector_type(8)))  _Float16 v8h;
typedef __attribute__((ext_vector_type(8)))  float    v8f;

#define TREES  10
#define BATCH  16384
#define INDIM  2048
#define NDEC   255
#define NDECP  256
#define NLEAF  512
#define NCLS   1000
#define NCLSP  1024

// ---- workspace layout (bytes) ----
#define WS_XH    ((size_t)0)            // BATCH*INDIM*2        = 67,108,864
#define WS_WDT   ((size_t)67108864)     // TREES*NDECP*INDIM*2  = 10,485,760
#define WS_LPT   ((size_t)78118912)     // TREES*NCLSP*NLEAF*2  = 10,485,760
#define WS_LPROB ((size_t)88604672)     // TREES*BATCH*NLEAF*2  = 167,772,160

#if defined(__HIP_DEVICE_COMPILE__) && __has_builtin(__builtin_amdgcn_tensor_load_to_lds) && __has_builtin(__builtin_amdgcn_s_wait_tensorcnt)
#define HAVE_TDM 1
#else
#define HAVE_TDM 0
#endif

#if HAVE_TDM
typedef unsigned int u32x4 __attribute__((ext_vector_type(4)));
typedef int          i32x8 __attribute__((ext_vector_type(8)));
typedef int          i32x4 __attribute__((ext_vector_type(4)));

// 2D Tensor-DMA load: tile [tile_d1 rows x tile_d0 elems] of 2-byte data from a
// row-major tensor (row pitch d0_stride elems) into LDS at byte offset lds_off.
// D# bitfields per CDNA5 ISA ch. 8 (group0: count/lds/global/type; group1: dims).
// clang-23 builtin form: (g0 u32x4, g1 i32x8, g2 i32x4, g3 i32x4, g4 i32x8, cpol)
// where g2/g3 (dims 2..4) and g4 (VADDR4, unused per ISA) are zero for 2D tiles.
static __device__ __forceinline__ void tdm_load_2d(unsigned lds_off,
                                                   const _Float16* gptr,
                                                   unsigned tensor_d0, unsigned tensor_d1,
                                                   unsigned d0_stride,
                                                   unsigned tile_d0, unsigned tile_d1) {
    unsigned long long ga = (unsigned long long)(size_t)gptr;
    u32x4 g0;
    g0[0] = 1u;                                                  // count=1, user mode
    g0[1] = lds_off;                                             // lds_addr (bytes)
    g0[2] = (unsigned)ga;                                        // global_addr[31:0]
    g0[3] = (unsigned)((ga >> 32) & 0x01ffffffu) | 0x80000000u;  // [56:32] | type=2
    i32x8 g1;
    g1[0] = (int)(1u << 16);                                     // data_size=1 (2 bytes)
    g1[1] = (int)((tensor_d0 & 0xffffu) << 16);                  // tensor_dim0[15:0]
    g1[2] = (int)((tensor_d0 >> 16) | ((tensor_d1 & 0xffffu) << 16));
    g1[3] = (int)((tensor_d1 >> 16) | (tile_d0 << 16));          // tile_dim0
    g1[4] = (int)(tile_d1 & 0xffffu);                            // tile_dim1 (tile_dim2=0)
    g1[5] = (int)d0_stride;                                      // tensor_dim0_stride[31:0]
    g1[6] = 0;
    g1[7] = 0;
    i32x4 gz4 = {0, 0, 0, 0};
    i32x8 gz8 = {0, 0, 0, 0, 0, 0, 0, 0};
    __builtin_amdgcn_tensor_load_to_lds(g0, g1, gz4, gz4, gz8, 0);
}
#endif

// A fragment (16x32 f16): per-lane halfs 0..7 at K + (hi?8:0), halfs 8..15 at +16
static __device__ __forceinline__ v16h load_afrag(const _Float16* p) {
    v8h lo = *(const v8h*)(p);
    v8h hi = *(const v8h*)(p + 16);
    return __builtin_shufflevector(lo, hi, 0,1,2,3,4,5,6,7,8,9,10,11,12,13,14,15);
}
// B fragment (32x16 f16): halfs 0..7 at K + (hi?8:0), halfs 8..15 at +16 (lane>=16 at +8 already applied)
static __device__ __forceinline__ v16h load_bfrag(const _Float16* p) {
    v8h lo = *(const v8h*)(p);
    v8h hi = *(const v8h*)(p + 16);
    return __builtin_shufflevector(lo, hi, 0,1,2,3,4,5,6,7,8,9,10,11,12,13,14,15);
}

// ---------------- x (f32) -> f16 ----------------
__global__ __launch_bounds__(256) void k_cvt_x(const float* __restrict__ x,
                                               _Float16* __restrict__ xh) {
    size_t i = (size_t)blockIdx.x * 256 + threadIdx.x;
    if (i < (size_t)BATCH * INDIM) xh[i] = (_Float16)x[i];
}

// ------- w_d [T,INDIM,NDEC] f32 -> wdT [T,NDECP,INDIM] f16 (transpose + pad) -------
__global__ __launch_bounds__(256) void k_cvt_wd(const float* __restrict__ wd,
                                                _Float16* __restrict__ wdT) {
    size_t i = (size_t)blockIdx.x * 256 + threadIdx.x;
    if (i >= (size_t)TREES * NDECP * INDIM) return;
    int t = (int)(i / ((size_t)NDECP * INDIM));
    int r = (int)(i % ((size_t)NDECP * INDIM));
    int n = r / INDIM;
    int k = r % INDIM;
    float v = (n < NDEC) ? wd[((size_t)t * INDIM + k) * NDEC + n] : 0.0f;
    wdT[i] = (_Float16)v;
}

// ------- softmax(w_l[T,NLEAF,NCLS]) -> leaf_pT [T,NCLSP,NLEAF] f16 (transposed, padded) -------
__global__ __launch_bounds__(256) void k_softmax_wl(const float* __restrict__ wl,
                                                    _Float16* __restrict__ lpt) {
    __shared__ float red[256];
    const int tid = threadIdx.x;
    const int row = blockIdx.x;           // t*NLEAF + l
    const int t = row >> 9;
    const int l = row & (NLEAF - 1);
    const float* src = wl + (size_t)row * NCLS;

    float mx = -3.4e38f;
    for (int c = tid; c < NCLS; c += 256) mx = fmaxf(mx, src[c]);
    red[tid] = mx; __syncthreads();
    for (int s = 128; s > 0; s >>= 1) { if (tid < s) red[tid] = fmaxf(red[tid], red[tid + s]); __syncthreads(); }
    mx = red[0]; __syncthreads();

    float sum = 0.0f;
    for (int c = tid; c < NCLS; c += 256) sum += __expf(src[c] - mx);
    red[tid] = sum; __syncthreads();
    for (int s = 128; s > 0; s >>= 1) { if (tid < s) red[tid] += red[tid + s]; __syncthreads(); }
    float inv = 1.0f / red[0];

    for (int c = tid; c < NCLSP; c += 256) {
        float v = (c < NCLS) ? __expf(src[c] - mx) * inv : 0.0f;
        lpt[((size_t)t * NCLSP + c) * NLEAF + l] = (_Float16)v;
    }
}

// ------- fused GEMM1 (x @ w_d[t]) + sigmoid + leaf products -> lprob [T,BATCH,NLEAF] f16 -----
// Block = 8 waves = 128 rows. B k-slice (256x32 f16, 16 KB) is TDM double-buffered
// in LDS and shared by all 8 waves; each wave owns a 16x256 tile in 16 f32 accumulators.
// After the k-loop the same 64 KB LDS is reused for the sigmoid tile (f16) feeding the
// closed-form leaf path products: node_n = 2^n-1 + (lp>>(8-n)), side = bit(7-n) of lp.
__global__ __launch_bounds__(256) void k_gemm1_leaf(const _Float16* __restrict__ xh,
                                                    const _Float16* __restrict__ wdT,
                                                    _Float16* __restrict__ lprob) {
    __shared__ __align__(16) char smem[65536];   // [0,32K): 2 x 16KB B buffers; then pt f16
    const int wv   = threadIdx.x >> 5;
    const int lane = threadIdx.x & 31;
    const int lo16 = lane & 15;
    const int hi   = lane >> 4;
    const int t     = blockIdx.y;
    const int mtile = blockIdx.x * 128 + wv * 16;

    v8f acc[16] = {};
    const _Float16* arow = xh + (size_t)(mtile + lo16) * INDIM + hi * 8;
    const _Float16* bsrc = wdT + (size_t)t * NDECP * INDIM;   // [256][2048]

    // preload k-step 0 into buffer 0
#if HAVE_TDM
    if (wv == 0) {
        tdm_load_2d((unsigned)(size_t)smem, bsrc, INDIM, NDECP, INDIM, 32, NDECP);
        __builtin_amdgcn_s_wait_tensorcnt(0);
    }
#else
    {   // 256 threads x one 64B row each
        const _Float16* s = bsrc + (size_t)threadIdx.x * INDIM;
        v8h* d = (v8h*)smem + threadIdx.x * 4;
        d[0] = *(const v8h*)(s);      d[1] = *(const v8h*)(s + 8);
        d[2] = *(const v8h*)(s + 16); d[3] = *(const v8h*)(s + 24);
    }
#endif
    __syncthreads();

    for (int kk = 0; kk < INDIM / 32; ++kk) {
        const int kb  = kk * 32;
        const int cur = kk & 1;
        if (kk + 1 < INDIM / 32) {
#if HAVE_TDM
            if (wv == 0)
                tdm_load_2d((unsigned)(size_t)(smem + ((cur ^ 1) << 14)),
                            bsrc + kb + 32, INDIM, NDECP, INDIM, 32, NDECP);
#else
            const _Float16* s = bsrc + (size_t)threadIdx.x * INDIM + kb + 32;
            v8h* d = (v8h*)(smem + ((cur ^ 1) << 14)) + threadIdx.x * 4;
            d[0] = *(const v8h*)(s);      d[1] = *(const v8h*)(s + 8);
            d[2] = *(const v8h*)(s + 16); d[3] = *(const v8h*)(s + 24);
#endif
        }
        v16h a = load_afrag(arow + kb);
        const _Float16* bufc = (const _Float16*)(smem + (cur << 14));  // [256 rows][32 halfs]
#pragma unroll
        for (int n = 0; n < 16; ++n) {
            v16h b = load_bfrag(bufc + (n * 16 + lo16) * 32 + hi * 8);
            acc[n] = __builtin_amdgcn_wmma_f32_16x16x32_f16(
                false, a, false, b, (short)0, acc[n], false, false);
        }
#if HAVE_TDM
        if (wv == 0) __builtin_amdgcn_s_wait_tensorcnt(0);
#endif
        __syncthreads();
    }

    // sigmoid -> LDS tile (reuse smem): f16 [8 waves][16 rows][256 decisions]
    _Float16* pt = (_Float16*)smem + wv * (16 * NDECP);
#pragma unroll
    for (int n = 0; n < 16; ++n) {
#pragma unroll
        for (int r = 0; r < 8; ++r) {
            float v = acc[n][r];
            pt[(r + hi * 8) * NDECP + (n * 16 + lo16)] =
                (_Float16)(1.0f / (1.0f + __expf(-v)));
        }
    }
    __syncthreads();

    // leaf path products (even/odd leaf pairs are identical; bit 0 unused)
    for (int idx = lane; idx < 16 * 256; idx += 32) {
        const int m  = idx >> 8;
        const int lp = idx & 255;
        float prod = 1.0f;
#pragma unroll
        for (int n = 0; n < 8; ++n) {
            const int node = (1 << n) - 1 + (lp >> (8 - n));
            const int side = (lp >> (7 - n)) & 1;
            const float p = (float)pt[m * NDECP + node];
            prod *= side ? (1.0f - p) : p;
        }
        union { _Float16 h[2]; unsigned u; } u2;
        u2.h[0] = (_Float16)prod; u2.h[1] = (_Float16)prod;
        *(unsigned*)(lprob + ((size_t)t * BATCH + mtile + m) * NLEAF + 2 * lp) = u2.u;
    }
}

// ------- GEMM2: out[b,c] = 0.1 * sum_t lprob[t,b,:] @ leaf_pT[t,c,:]  (K = 10*512) -------
// Block = 8 waves = 128 rows sharing one 64-column B slice; B k-step tile (64x32 f16,
// 4 KB) TDM double-buffered in LDS. Wave tile 16x64 (4 f32 accumulators).
__global__ __launch_bounds__(256) void k_gemm2(const _Float16* __restrict__ lprob,
                                               const _Float16* __restrict__ lpt,
                                               float* __restrict__ out) {
    __shared__ __align__(16) char smem[8192];    // 2 x 4KB B buffers
    const int wv   = threadIdx.x >> 5;
    const int lane = threadIdx.x & 31;
    const int lo16 = lane & 15;
    const int hi   = lane >> 4;
    const int mrow  = blockIdx.x * 128 + wv * 16;
    const int ntile = blockIdx.y * 64;

    v8f acc[4] = {};

#if HAVE_TDM
    if (wv == 0) {
        tdm_load_2d((unsigned)(size_t)smem, lpt + (size_t)ntile * NLEAF,
                    NLEAF, NCLSP, NLEAF, 32, 64);
        __builtin_amdgcn_s_wait_tensorcnt(0);
    }
#else
    {   // 64 rows x 64B = 4KB; 256 threads x 16B
        const int r = threadIdx.x >> 2, ch = threadIdx.x & 3;
        *((v8h*)smem + threadIdx.x) =
            *(const v8h*)(lpt + ((size_t)ntile + r) * NLEAF + ch * 8);
    }
#endif
    __syncthreads();

    for (int it = 0; it < TREES * (NLEAF / 32); ++it) {
        const int t   = it >> 4;
        const int kb  = (it & 15) * 32;
        const int cur = it & 1;
        if (it + 1 < TREES * (NLEAF / 32)) {
            const int t2  = (it + 1) >> 4;
            const int kb2 = ((it + 1) & 15) * 32;
#if HAVE_TDM
            if (wv == 0)
                tdm_load_2d((unsigned)(size_t)(smem + ((cur ^ 1) << 12)),
                            lpt + ((size_t)t2 * NCLSP + ntile) * NLEAF + kb2,
                            NLEAF, NCLSP, NLEAF, 32, 64);
#else
            const int r = threadIdx.x >> 2, ch = threadIdx.x & 3;
            *((v8h*)(smem + ((cur ^ 1) << 12)) + threadIdx.x) =
                *(const v8h*)(lpt + ((size_t)t2 * NCLSP + ntile + r) * NLEAF + kb2 + ch * 8);
#endif
        }
        v16h a = load_afrag(lprob + ((size_t)t * BATCH + mrow + lo16) * NLEAF + kb + hi * 8);
        const _Float16* bufc = (const _Float16*)(smem + (cur << 12));  // [64 rows][32 halfs]
#pragma unroll
        for (int n = 0; n < 4; ++n) {
            v16h b = load_bfrag(bufc + (n * 16 + lo16) * 32 + hi * 8);
            acc[n] = __builtin_amdgcn_wmma_f32_16x16x32_f16(
                false, a, false, b, (short)0, acc[n], false, false);
        }
#if HAVE_TDM
        if (wv == 0) __builtin_amdgcn_s_wait_tensorcnt(0);
#endif
        __syncthreads();
    }

#pragma unroll
    for (int n = 0; n < 4; ++n) {
        const int c = ntile + n * 16 + lo16;
        if (c < NCLS) {
#pragma unroll
            for (int r = 0; r < 8; ++r)
                out[(size_t)(mrow + r + hi * 8) * NCLS + c] = acc[n][r] * 0.1f;
        }
    }
}

extern "C" void kernel_launch(void* const* d_in, const int* in_sizes, int n_in,
                              void* d_out, int out_size, void* d_ws, size_t ws_size,
                              hipStream_t stream) {
    const float* x  = (const float*)d_in[0];   // [16384, 2048]
    const float* wd = (const float*)d_in[1];   // [10, 2048, 255]
    const float* wl = (const float*)d_in[2];   // [10, 512, 1000]
    float* out = (float*)d_out;                // [16384, 1000]

    char* ws = (char*)d_ws;
    _Float16* xh    = (_Float16*)(ws + WS_XH);
    _Float16* wdT   = (_Float16*)(ws + WS_WDT);
    _Float16* lpt   = (_Float16*)(ws + WS_LPT);
    _Float16* lprob = (_Float16*)(ws + WS_LPROB);

    {   // x -> f16
        size_t n = (size_t)BATCH * INDIM;
        k_cvt_x<<<dim3((unsigned)((n + 255) / 256)), dim3(256), 0, stream>>>(x, xh);
    }
    {   // w_d -> transposed padded f16
        size_t n = (size_t)TREES * NDECP * INDIM;
        k_cvt_wd<<<dim3((unsigned)((n + 255) / 256)), dim3(256), 0, stream>>>(wd, wdT);
    }
    {   // softmax(w_l) -> transposed padded f16
        k_softmax_wl<<<dim3(TREES * NLEAF), dim3(256), 0, stream>>>(wl, lpt);
    }
    {   // fused GEMM1 + sigmoid + leaf products (8 waves/block, TDM-staged B)
        k_gemm1_leaf<<<dim3(BATCH / 128, TREES), dim3(256), 0, stream>>>(xh, wdT, lprob);
    }
    {   // GEMM2 + scale (8 waves/block, TDM-staged B)
        k_gemm2<<<dim3(BATCH / 128, NCLSP / 64), dim3(256), 0, stream>>>(lprob, lpt, out);
    }
}
